// DSVTBackboneMAE_74981539053617
// MI455X (gfx1250) — matrix operand order repaired and checked by reading, not compiled, not run on hardware
//
#include <hip/hip_runtime.h>
#include <hip/hip_bf16.h>
#include <math.h>

#define C_DIM 192
#define L_SET 48
#define S_NUM 4096
#define N_VOX 196608
#define NHEADS 8
#define HD 24
#define DFF 384

typedef __attribute__((ext_vector_type(16))) __bf16 v16bf;
typedef __attribute__((ext_vector_type(8)))  float  v8f;

static __device__ __forceinline__ __bf16 f2bf(float f) {
  union { float f; unsigned u; } a; a.f = f;
  unsigned r = a.u + 0x7FFFu + ((a.u >> 16) & 1u);   // round-to-nearest-even
  unsigned short h = (unsigned short)(r >> 16);
  return __builtin_bit_cast(__bf16, h);
}

static __device__ __forceinline__ float wave_sum(float v) {
#pragma unroll
  for (int off = 16; off >= 1; off >>= 1) v += __shfl_xor(v, off, 32);
  return v;
}
static __device__ __forceinline__ float wave_max(float v) {
#pragma unroll
  for (int off = 16; off >= 1; off >>= 1) v = fmaxf(v, __shfl_xor(v, off, 32));
  return v;
}

// Async DMA: 16 bytes global -> LDS, per-lane addressing (GVS mode).
// ISA 15.18.3 op 98: GLOBAL_LOAD_ASYNC_TO_LDS_B128, tracked by ASYNCcnt.
static __device__ __forceinline__ void async_load_b128(unsigned lds_addr,
                                                       unsigned long long base,
                                                       unsigned voff) {
  asm volatile("global_load_async_to_lds_b128 %0, %1, %2 offset:0"
               :: "v"(lds_addr), "v"(voff), "s"(base)
               : "memory");
}
static __device__ __forceinline__ void wait_async0() {
  asm volatile("s_wait_asynccnt 0" ::: "memory");
}
static __device__ __forceinline__ unsigned lds_addr_of(const void* p) {
  return (unsigned)(unsigned long long)p;   // low 32 bits of LDS aperture addr
}

// fp32 -> bf16, transposed: in is rows x cols row-major, out is cols x rows.
__global__ void cvt_bf16_t(const float* __restrict__ in, __bf16* __restrict__ out,
                           int rows, int cols) {
  int i = blockIdx.x * 256 + threadIdx.x;
  if (i < rows * cols) {
    int r = i / cols, c = i - r * cols;
    out[c * rows + r] = f2bf(in[r * cols + c]);
  }
}

// ---------------------------------------------------------------------------
// Kernel 1: per-set attention (async gather -> QKV -> scores -> softmax ->
// ctx -> out-proj -> scatter + residual + LN1 into xbuf). One block per set.
// LDS layout (bytes):
//   [0,      36864)  qk_in bf16 48x192 | feat bf16 48x192  (later: attn bf16 8x48x48)
//   [36864,  55296)  Q bf16 48x192
//   [55296,  73728)  K^T bf16 192x48
//   [73728,  92160)  V bf16 48x192
//   [92160, 165888)  phase1: src f32 stage @92160, pos f32 stage @129024
//                    phase3: scores f32 8x48x48
//                    phase4+: ctx bf16 48x192 @92160, out f32 48x192 @110592
// ---------------------------------------------------------------------------
__global__ void __launch_bounds__(256)
dsvt_attn_kernel(const float* __restrict__ src, const float* __restrict__ pos,
                 const int* __restrict__ vinds, const unsigned char* __restrict__ kmask,
                 const float* __restrict__ b_qkv, const float* __restrict__ b_out,
                 const float* __restrict__ ln1_g, const float* __restrict__ ln1_b,
                 const __bf16* __restrict__ wqkv_t, const __bf16* __restrict__ wout_t,
                 float* __restrict__ xbuf) {
  extern __shared__ char smem[];
  __bf16* qkb   = (__bf16*)(smem + 0);
  __bf16* featb = (__bf16*)(smem + 18432);
  __bf16* attnb = (__bf16*)(smem + 0);        // overlays qkb/featb after phase 2
  __bf16* qb    = (__bf16*)(smem + 36864);
  __bf16* ktb   = (__bf16*)(smem + 55296);    // transposed K: [c][l]
  __bf16* vb    = (__bf16*)(smem + 73728);
  float*  stg_s = (float*)(smem + 92160);     // staged src rows (f32)
  float*  stg_p = (float*)(smem + 129024);    // staged pos rows (f32)
  float*  scf   = (float*)(smem + 92160);     // 8 heads x 48 x 48
  __bf16* ctxb  = (__bf16*)(smem + 92160);    // overlays scf after barrier
  float*  outf  = (float*)(smem + 110592);    // overlays scf tail

  const int tid  = threadIdx.x;
  const int lane = tid & 31;
  const int w    = tid >> 5;       // wave id 0..7  (== head id in attn phases)
  const int half = lane >> 4;      // 0|1
  const int mrow = lane & 15;
  const int s    = blockIdx.x;

  // -------- phase 1a: async DMA gather of src/pos rows into LDS (f32) ------
  // 48 rows x 768 B = 48 granules of 16 B per row; 2304 granules per tensor.
  {
    const unsigned lds0 = lds_addr_of(smem);
    for (int g = tid; g < L_SET * 48; g += 256) {
      int row = g / 48;
      int off = (g - row * 48) * 16;               // byte offset within row
      int vx  = vinds[s * L_SET + row];
      unsigned goff = (unsigned)vx * 768u + (unsigned)off;
      async_load_b128(lds0 + 92160u + (unsigned)(row * 768 + off),
                      (unsigned long long)src, goff);
      async_load_b128(lds0 + 129024u + (unsigned)(row * 768 + off),
                      (unsigned long long)pos, goff);
    }
    wait_async0();
  }
  __syncthreads();

  // -------- phase 1b: convert to bf16 feat / qk_in --------
  for (int i = tid; i < L_SET * C_DIM; i += 256) {
    float fv = stg_s[i];
    float pv = stg_p[i];
    featb[i] = f2bf(fv);
    qkb[i]   = f2bf(fv + pv);
  }
  __syncthreads();

  // -------- phase 2: QKV projection, 48x576 = 3x36 output tiles --------
  for (int t = w; t < 108; t += 8) {
    int mt = t / 36, nt = t % 36;
    int n0 = nt * 16;                               // global out column 0..575
    const __bf16* asrc = (n0 < 2 * C_DIM) ? qkb : featb;
    v8f acc = {};
#pragma unroll
    for (int kk = 0; kk < 6; ++kk) {
      v16bf a, b;
#pragma unroll
      for (int e = 0; e < 16; ++e) {
        int kl = (e & 7) + ((e >> 3) << 4) + (half << 3);
        a[e] = asrc[(mt * 16 + mrow) * C_DIM + kk * 32 + kl];
        b[e] = wqkv_t[(kk * 32 + kl) * 576 + n0 + mrow];   // k-major weights
      }
      acc = __builtin_amdgcn_wmma_f32_16x16x32_bf16(false, a, false, b,
                                                    (short)0, acc, false, false);
    }
    int ncol = n0 % C_DIM;
    if (n0 < C_DIM) {                                      // Q, row-major
#pragma unroll
      for (int r = 0; r < 8; ++r) {
        int m = mt * 16 + r + (half << 3);
        qb[m * C_DIM + ncol + mrow] = f2bf(acc[r] + b_qkv[n0 + mrow]);
      }
    } else if (n0 < 2 * C_DIM) {                           // K, transposed
#pragma unroll
      for (int r = 0; r < 8; ++r) {
        int m = mt * 16 + r + (half << 3);
        ktb[(ncol + mrow) * L_SET + m] = f2bf(acc[r] + b_qkv[n0 + mrow]);
      }
    } else {                                               // V, row-major
#pragma unroll
      for (int r = 0; r < 8; ++r) {
        int m = mt * 16 + r + (half << 3);
        vb[m * C_DIM + ncol + mrow] = f2bf(acc[r] + b_qkv[n0 + mrow]);
      }
    }
  }
  __syncthreads();

  // -------- phase 3: scores + softmax; wave w owns head h=w --------
  {
    const int h = w;
    const float scale = 0.20412414523193154f;   // 1/sqrt(24)
    for (int mt = 0; mt < 3; ++mt)
      for (int nt = 0; nt < 3; ++nt) {
        v16bf a, b;
#pragma unroll
        for (int e = 0; e < 16; ++e) {
          int kl = (e & 7) + ((e >> 3) << 4) + (half << 3);   // 0..31, pad >=24
          a[e] = (kl < HD) ? qb[(mt * 16 + mrow) * C_DIM + h * HD + kl] : f2bf(0.f);
          b[e] = (kl < HD) ? ktb[(h * HD + kl) * L_SET + nt * 16 + mrow] : f2bf(0.f);
        }
        v8f acc = {};
        acc = __builtin_amdgcn_wmma_f32_16x16x32_bf16(false, a, false, b,
                                                      (short)0, acc, false, false);
#pragma unroll
        for (int r = 0; r < 8; ++r) {
          int m = mt * 16 + r + (half << 3);
          scf[h * 2304 + m * 48 + nt * 16 + mrow] = acc[r] * scale;
        }
      }
    // softmax over 48 rows of this head; lane covers cols {lane, lane+32}
    for (int row = 0; row < L_SET; ++row) {
      float v1 = scf[h * 2304 + row * 48 + lane];
      if (kmask[s * L_SET + lane]) v1 = -1e9f;
      float v2 = -1e30f;
      if (lane < 16) {
        v2 = scf[h * 2304 + row * 48 + lane + 32];
        if (kmask[s * L_SET + lane + 32]) v2 = -1e9f;
      }
      float mx = wave_max(fmaxf(v1, v2));
      float e1 = __expf(v1 - mx);
      float e2 = (lane < 16) ? __expf(v2 - mx) : 0.f;
      float inv = 1.f / wave_sum(e1 + e2);
      attnb[h * 2304 + row * 48 + lane] = f2bf(e1 * inv);
      if (lane < 16) attnb[h * 2304 + row * 48 + lane + 32] = f2bf(e2 * inv);
    }
  }
  __syncthreads();   // scores region dead -> reuse as ctx/out

  // -------- phase 4: ctx = attn @ V (per head), K=48 padded to 64 --------
  {
    const int h = w;
    for (int mt = 0; mt < 3; ++mt)
      for (int nt2 = 0; nt2 < 2; ++nt2) {     // hd=24 padded to 32
        v8f acc = {};
#pragma unroll
        for (int ks = 0; ks < 2; ++ks) {
          v16bf a, b;
#pragma unroll
          for (int e = 0; e < 16; ++e) {
            int kl = ks * 32 + (e & 7) + ((e >> 3) << 4) + (half << 3);
            a[e] = (kl < 48) ? attnb[h * 2304 + (mt * 16 + mrow) * 48 + kl] : f2bf(0.f);
            int kr = kl;
            int nl = nt2 * 16 + mrow;
            b[e] = (kr < 48 && nl < HD) ? vb[kr * C_DIM + h * HD + nl] : f2bf(0.f);
          }
          acc = __builtin_amdgcn_wmma_f32_16x16x32_bf16(false, a, false, b,
                                                        (short)0, acc, false, false);
        }
        int nl = nt2 * 16 + mrow;
        if (nl < HD) {
#pragma unroll
          for (int r = 0; r < 8; ++r) {
            int m = mt * 16 + r + (half << 3);
            ctxb[m * C_DIM + h * HD + nl] = f2bf(acc[r]);
          }
        }
      }
  }
  __syncthreads();

  // -------- phase 5: out projection 48x192, 3x12 tiles --------
  for (int t = w; t < 36; t += 8) {
    int mt = t / 12, nt = t % 12;
    int n0 = nt * 16;
    v8f acc = {};
#pragma unroll
    for (int kk = 0; kk < 6; ++kk) {
      v16bf a, b;
#pragma unroll
      for (int e = 0; e < 16; ++e) {
        int kl = (e & 7) + ((e >> 3) << 4) + (half << 3);
        a[e] = ctxb[(mt * 16 + mrow) * C_DIM + kk * 32 + kl];
        b[e] = wout_t[(kk * 32 + kl) * C_DIM + n0 + mrow];
      }
      acc = __builtin_amdgcn_wmma_f32_16x16x32_bf16(false, a, false, b,
                                                    (short)0, acc, false, false);
    }
#pragma unroll
    for (int r = 0; r < 8; ++r) {
      int m = mt * 16 + r + (half << 3);
      outf[m * C_DIM + n0 + mrow] = acc[r] + b_out[n0 + mrow];
    }
  }
  __syncthreads();

  // -------- phase 6: scatter + residual + LN1 -> xbuf (voxel order) --------
  for (int i = 0; i < 6; ++i) {
    int row = w * 6 + i;
    int vx = vinds[s * L_SET + row];
    float vals[6]; float s1 = 0.f, s2 = 0.f;
#pragma unroll
    for (int j = 0; j < 6; ++j) {
      int c = lane + 32 * j;
      float t = src[vx * C_DIM + c] + outf[row * C_DIM + c];
      vals[j] = t; s1 += t; s2 += t * t;
    }
    s1 = wave_sum(s1); s2 = wave_sum(s2);
    float mean = s1 * (1.f / 192.f);
    float var  = s2 * (1.f / 192.f) - mean * mean;
    float rstd = rsqrtf(var + 1e-5f);
#pragma unroll
    for (int j = 0; j < 6; ++j) {
      int c = lane + 32 * j;
      xbuf[vx * C_DIM + c] = (vals[j] - mean) * rstd * ln1_g[c] + ln1_b[c];
    }
  }
}

// ---------------------------------------------------------------------------
// Kernel 2: FFN + LN2 + encoder residual + LN3. 64 rows per block.
// LDS: xb bf16 64x192 @0 (24576) | xf f32 64x192 @24576 (49152, async DMA)
//      hb bf16 64x384 @73728 (49152) | yf f32 64x192 @122880 (49152)
// ---------------------------------------------------------------------------
__global__ void __launch_bounds__(256)
dsvt_ffn_kernel(const float* __restrict__ src,
                const float* __restrict__ b1, const float* __restrict__ b2,
                const float* __restrict__ ln2_g, const float* __restrict__ ln2_b,
                const float* __restrict__ ln3_g, const float* __restrict__ ln3_b,
                const __bf16* __restrict__ w1_t, const __bf16* __restrict__ w2_t,
                const float* __restrict__ xbuf, float* __restrict__ out) {
  extern __shared__ char smem[];
  __bf16* xb = (__bf16*)(smem);
  float*  xf = (float*)(smem + 24576);
  __bf16* hb = (__bf16*)(smem + 73728);
  float*  yf = (float*)(smem + 122880);

  const int tid  = threadIdx.x;
  const int lane = tid & 31;
  const int w    = tid >> 5;
  const int half = lane >> 4;
  const int mrow = lane & 15;
  const int r0   = blockIdx.x * 64;

  // async DMA the contiguous 64x192 f32 x-tile straight into LDS
  {
    const unsigned lds0 = lds_addr_of(smem);
    const unsigned long long base = (unsigned long long)(xbuf + (size_t)r0 * C_DIM);
    for (int i = tid; i < 3072; i += 256) {           // 49152 B / 16
      unsigned off = (unsigned)i * 16u;
      async_load_b128(lds0 + 24576u + off, base, off);
    }
    wait_async0();
  }
  __syncthreads();
  for (int i = tid; i < 64 * C_DIM; i += 256) xb[i] = f2bf(xf[i]);
  __syncthreads();

  // gemm1: (64x192)@(192x384) -> relu -> hb ; 4x24 tiles
  for (int t = w; t < 96; t += 8) {
    int mt = t / 24, nt = t % 24;
    int n0 = nt * 16;
    v8f acc = {};
#pragma unroll
    for (int kk = 0; kk < 6; ++kk) {
      v16bf a, b;
#pragma unroll
      for (int e = 0; e < 16; ++e) {
        int kl = (e & 7) + ((e >> 3) << 4) + (half << 3);
        a[e] = xb[(mt * 16 + mrow) * C_DIM + kk * 32 + kl];
        b[e] = w1_t[(kk * 32 + kl) * DFF + n0 + mrow];
      }
      acc = __builtin_amdgcn_wmma_f32_16x16x32_bf16(false, a, false, b,
                                                    (short)0, acc, false, false);
    }
#pragma unroll
    for (int r = 0; r < 8; ++r) {
      int m = mt * 16 + r + (half << 3);
      float val = acc[r] + b1[n0 + mrow];
      hb[m * DFF + n0 + mrow] = f2bf(fmaxf(val, 0.f));
    }
  }
  __syncthreads();

  // gemm2: (64x384)@(384x192) -> yf ; 4x12 tiles, K=12 steps
  for (int t = w; t < 48; t += 8) {
    int mt = t / 12, nt = t % 12;
    int n0 = nt * 16;
    v8f acc = {};
#pragma unroll
    for (int kk = 0; kk < 12; ++kk) {
      v16bf a, b;
#pragma unroll
      for (int e = 0; e < 16; ++e) {
        int kl = (e & 7) + ((e >> 3) << 4) + (half << 3);
        a[e] = hb[(mt * 16 + mrow) * DFF + kk * 32 + kl];
        b[e] = w2_t[(kk * 32 + kl) * C_DIM + n0 + mrow];
      }
      acc = __builtin_amdgcn_wmma_f32_16x16x32_bf16(false, a, false, b,
                                                    (short)0, acc, false, false);
    }
#pragma unroll
    for (int r = 0; r < 8; ++r) {
      int m = mt * 16 + r + (half << 3);
      yf[m * C_DIM + n0 + mrow] = acc[r] + b2[n0 + mrow];
    }
  }
  __syncthreads();

  // epilogue: LN2(x+ffn) then LN3(that + identity) -> out
  for (int i = 0; i < 8; ++i) {
    int row = w * 8 + i;
    int g = r0 + row;
    float t1[6]; float s1 = 0.f, s2 = 0.f;
#pragma unroll
    for (int j = 0; j < 6; ++j) {
      int c = lane + 32 * j;
      float v = xf[row * C_DIM + c] + yf[row * C_DIM + c];
      t1[j] = v; s1 += v; s2 += v * v;
    }
    s1 = wave_sum(s1); s2 = wave_sum(s2);
    float mean = s1 * (1.f / 192.f);
    float var  = s2 * (1.f / 192.f) - mean * mean;
    float rstd = rsqrtf(var + 1e-5f);
    float t2[6]; float u1 = 0.f, u2 = 0.f;
#pragma unroll
    for (int j = 0; j < 6; ++j) {
      int c = lane + 32 * j;
      float x2 = (t1[j] - mean) * rstd * ln2_g[c] + ln2_b[c];
      float v = x2 + src[g * C_DIM + c];
      t2[j] = v; u1 += v; u2 += v * v;
    }
    u1 = wave_sum(u1); u2 = wave_sum(u2);
    float mean2 = u1 * (1.f / 192.f);
    float var2  = u2 * (1.f / 192.f) - mean2 * mean2;
    float rstd2 = rsqrtf(var2 + 1e-5f);
#pragma unroll
    for (int j = 0; j < 6; ++j) {
      int c = lane + 32 * j;
      out[g * C_DIM + c] = (t2[j] - mean2) * rstd2 * ln3_g[c] + ln3_b[c];
    }
  }
}

// ---------------------------------------------------------------------------
extern "C" void kernel_launch(void* const* d_in, const int* in_sizes, int n_in,
                              void* d_out, int out_size, void* d_ws, size_t ws_size,
                              hipStream_t stream) {
  (void)in_sizes; (void)n_in; (void)out_size; (void)ws_size;
  const float* src  = (const float*)d_in[0];
  const float* pos  = (const float*)d_in[1];
  const float* wqkv = (const float*)d_in[2];
  const float* bqkv = (const float*)d_in[3];
  const float* wout = (const float*)d_in[4];
  const float* bout = (const float*)d_in[5];
  const float* w1   = (const float*)d_in[6];
  const float* b1   = (const float*)d_in[7];
  const float* w2   = (const float*)d_in[8];
  const float* b2   = (const float*)d_in[9];
  const float* ln1g = (const float*)d_in[10];
  const float* ln1b = (const float*)d_in[11];
  const float* ln2g = (const float*)d_in[12];
  const float* ln2b = (const float*)d_in[13];
  const float* ln3g = (const float*)d_in[14];
  const float* ln3b = (const float*)d_in[15];
  const int*   vinds = (const int*)d_in[16];
  const unsigned char* kmask = (const unsigned char*)d_in[17];

  char* ws = (char*)d_ws;
  __bf16* wqkv_t = (__bf16*)(ws + 0);        // 192 x 576
  __bf16* wout_t = (__bf16*)(ws + 221184);   // 192 x 192
  __bf16* w1_t   = (__bf16*)(ws + 294912);   // 192 x 384
  __bf16* w2_t   = (__bf16*)(ws + 442368);   // 384 x 192
  float*  xbuf   = (float*)(ws + 589824);    // N x 192 f32

  cvt_bf16_t<<<(576 * 192 + 255) / 256, 256, 0, stream>>>(wqkv, wqkv_t, 576, 192);
  cvt_bf16_t<<<(192 * 192 + 255) / 256, 256, 0, stream>>>(wout, wout_t, 192, 192);
  cvt_bf16_t<<<(384 * 192 + 255) / 256, 256, 0, stream>>>(w1, w1_t, 384, 192);
  cvt_bf16_t<<<(192 * 384 + 255) / 256, 256, 0, stream>>>(w2, w2_t, 192, 384);

  dsvt_attn_kernel<<<S_NUM, 256, 165888, stream>>>(
      src, pos, vinds, kmask, bqkv, bout, ln1g, ln1b, wqkv_t, wout_t, xbuf);

  dsvt_ffn_kernel<<<N_VOX / 64, 256, 172032, stream>>>(
      src, b1, b2, ln2g, ln2b, ln3g, ln3b, w1_t, w2_t, xbuf, (float*)d_out);
}
